// RevisedGAE_43499428774085
// MI455X (gfx1250) — matrix operand order repaired and checked by reading, not compiled, not run on hardware
//
#include <hip/hip_runtime.h>

// ---------------------------------------------------------------------------
// RevisedGAE on MI455X (gfx1250): GCN x2 + MLP head.
//   h1 = relu(dinv*(Adj @ (dinv*W1)) + 2*dinv^2*W1 + b1)               [8192x512]
//   M2 = h1 @ W2 ; h2 = relu(dinv*(Adj @ (dinv*M2)) + 2dinv^2*M2 + b2) [8192x128]
//   Z  = relu(h2 @ Wf1 + bf1) ; A = Z @ Wf2 + bf2
// Adj is binarized (exact in bf16: 0/1). All GEMMs: v_wmma_f32_16x16x32_bf16,
// fp32 accumulate, fused epilogues, double-buffered LDS staging fed by
// global_load_async_to_lds_b128 (ASYNCcnt) when the toolchain exposes it.
// ---------------------------------------------------------------------------

typedef __attribute__((ext_vector_type(16))) __bf16 v16bf;
typedef __attribute__((ext_vector_type(8)))  float  v8f;
typedef __attribute__((ext_vector_type(4)))  int    v4i;
typedef unsigned short u16;
typedef unsigned int   u32;

#define GLOBAL_AS __attribute__((address_space(1)))
#define LDS_AS    __attribute__((address_space(3)))

#if defined(__has_builtin)
#if __has_builtin(__builtin_amdgcn_global_load_async_to_lds_b128) && \
    __has_builtin(__builtin_amdgcn_s_wait_asynccnt)
#define USE_ASYNC 1
#endif
#endif
#ifndef USE_ASYNC
#define USE_ASYNC 0
#endif

static __device__ __forceinline__ u16 f2bf(float f) {
    u32 u = __float_as_uint(f);
    u32 r = u + 0x7FFFu + ((u >> 16) & 1u);   // round-to-nearest-even
    return (u16)(r >> 16);
}

// ---------------------------------------------------------------------------
// Pass 1: per-row nonzero count -> dinv = rsqrt(deg+2); binarize adj to bf16.
// ---------------------------------------------------------------------------
__global__ __launch_bounds__(256)
void k_deg_binarize(const float* __restrict__ adj, u16* __restrict__ adjb,
                    float* __restrict__ dinv, int n) {
    const int row = blockIdx.x;
    const int tid = threadIdx.x;
    const float4* arow = (const float4*)(adj + (size_t)row * n);
    u16* orow = adjb + (size_t)row * n;
    const u16 ONE = 0x3F80u;  // bf16 1.0
    int cnt = 0;
    for (int i = tid; i < n / 4; i += 256) {
        float4 v = arow[i];
        u16 o0 = (v.x != 0.f) ? ONE : 0; cnt += (v.x != 0.f);
        u16 o1 = (v.y != 0.f) ? ONE : 0; cnt += (v.y != 0.f);
        u16 o2 = (v.z != 0.f) ? ONE : 0; cnt += (v.z != 0.f);
        u16 o3 = (v.w != 0.f) ? ONE : 0; cnt += (v.w != 0.f);
        uint2 pk;
        pk.x = (u32)o0 | ((u32)o1 << 16);
        pk.y = (u32)o2 | ((u32)o3 << 16);
        *(uint2*)(orow + (size_t)i * 4) = pk;
    }
    __shared__ int red[256];
    red[tid] = cnt;
    __syncthreads();
    for (int s = 128; s > 0; s >>= 1) {
        if (tid < s) red[tid] += red[tid + s];
        __syncthreads();
    }
    if (tid == 0) dinv[row] = rsqrtf((float)red[0] + 2.0f);
}

// out[r,c] = bf16(dinv[r] * W[r,c])
__global__ void k_scale_rows_bf16(const float* __restrict__ W, const float* __restrict__ dinv,
                                  u16* __restrict__ out, int rows, int cols) {
    size_t total = (size_t)rows * cols;
    for (size_t idx = (size_t)blockIdx.x * blockDim.x + threadIdx.x; idx < total;
         idx += (size_t)gridDim.x * blockDim.x) {
        int r = (int)(idx / cols);
        out[idx] = f2bf(dinv[r] * W[idx]);
    }
}

__global__ void k_f32_to_bf16(const float* __restrict__ in, u16* __restrict__ out, int n) {
    for (int idx = blockIdx.x * blockDim.x + threadIdx.x; idx < n;
         idx += gridDim.x * blockDim.x)
        out[idx] = f2bf(in[idx]);
}

// ---------------------------------------------------------------------------
// Tiled bf16 WMMA GEMM: C[M,N] = A[M,K] @ B[K,N], fp32 accumulate.
// Block = 256 threads = 8 waves; tile 128x128, K-step 32, double-buffered LDS.
// Wave (wm = wave&3, wn = wave>>2) owns rows wm*32..+32, cols wn*64..+64
// => 2x4 accumulator tiles of v8f.
// A staged row-major in LDS (async direct-to-LDS b128 when available);
// B staged TRANSPOSED (Bs[n][k]) so both fragment builds are contiguous
// ds_load_b128 pairs matching the ISA 16-bit A(16x32)/B(32x16) wave32 layouts.
// All launch dims are exact tile multiples -> straight-line epilogue, no guards.
// ---------------------------------------------------------------------------
#define BM 128
#define BN 128
#define KT 32
#define LPAD 8   // LDS row padding (u16): row stride 80B, keeps 16B alignment

template <int EPI>
__global__ __launch_bounds__(256)
void k_gemm_bf16(const u16* __restrict__ A, const u16* __restrict__ B,
                 float* __restrict__ outF, u16* __restrict__ outBF,
                 const float* __restrict__ dinv, const float* __restrict__ Mdiag,
                 const float* __restrict__ bias,
                 int K, int lda, int ldb, int ldd) {
    __shared__ u16 As[2][BM][KT + LPAD];   // row-major:  As[b][m][k]
    __shared__ u16 Bs[2][BN][KT + LPAD];   // transposed: Bs[b][n][k]

    const int tid  = threadIdx.x;
    const int wave = tid >> 5;
    const int lane = tid & 31;
    const int wm   = wave & 3;    // row-tile group (32 rows)
    const int wn   = wave >> 2;   // col-tile group (64 cols)
    const int rb   = blockIdx.y * BM;
    const int cb   = blockIdx.x * BN;
    const int mrow = lane & 15;
    const int sel  = lane >> 4;

    v8f acc[2][4];
#pragma unroll
    for (int i = 0; i < 2; ++i)
#pragma unroll
        for (int j = 0; j < 4; ++j) acc[i][j] = {};

    union Frag { uint4 q[2]; v16bf v; };

    // --- staging helpers -------------------------------------------------
    auto stageA = [&](int k0, int buf) {
        const int r = tid >> 1, part = tid & 1;
        const u16* src = A + (size_t)(rb + r) * lda + k0 + part * 16;
        u16* dst = &As[buf][r][part * 16];
#if USE_ASYNC
        __builtin_amdgcn_global_load_async_to_lds_b128(
            (GLOBAL_AS v4i*)src, (LDS_AS v4i*)dst, 0, 0);
        __builtin_amdgcn_global_load_async_to_lds_b128(
            (GLOBAL_AS v4i*)(src + 8), (LDS_AS v4i*)(dst + 8), 0, 0);
#else
        uint4 q0 = *(const uint4*)(src);
        uint4 q1 = *(const uint4*)(src + 8);
        *(uint4*)dst       = q0;
        *(uint4*)(dst + 8) = q1;
#endif
    };
    auto stageB = [&](int k0, int buf) {
        // element-level transpose -> must go through registers
        const int k = tid >> 3, seg = tid & 7;
        const u16* src = B + (size_t)(k0 + k) * ldb + cb + seg * 16;
        union { uint4 q[2]; u16 s[16]; } t;
        t.q[0] = *(const uint4*)(src);
        t.q[1] = *(const uint4*)(src + 8);
#pragma unroll
        for (int j = 0; j < 16; ++j) Bs[buf][seg * 16 + j][k] = t.s[j];
    };

    // --- prologue: fill buffer 0 ----------------------------------------
    stageA(0, 0);
    stageB(0, 0);
#if USE_ASYNC
    __builtin_amdgcn_s_wait_asynccnt(0);
#endif
    __syncthreads();

    int buf = 0;
    for (int k0 = 0; k0 < K; k0 += KT) {
        // stage next tile into the other buffer while computing this one
        const int kn = k0 + KT;
        if (kn < K) {
            stageA(kn, buf ^ 1);
            stageB(kn, buf ^ 1);
        }

        // --- build fragments from LDS (contiguous b128 pairs) ---
        Frag af[2], bf[4];
#pragma unroll
        for (int am = 0; am < 2; ++am) {
            const u16* p = &As[buf][wm * 32 + am * 16 + mrow][0];
            af[am].q[0] = *(const uint4*)(p + sel * 8);        // K = sel*8 .. +7
            af[am].q[1] = *(const uint4*)(p + 16 + sel * 8);   // K = 16+sel*8 .. +7
        }
#pragma unroll
        for (int bn = 0; bn < 4; ++bn) {
            const u16* p = &Bs[buf][wn * 64 + bn * 16 + mrow][0];
            bf[bn].q[0] = *(const uint4*)(p + sel * 8);        // K = sel*8 .. +7
            bf[bn].q[1] = *(const uint4*)(p + 16 + sel * 8);   // K = 16+sel*8 .. +7
        }

#pragma unroll
        for (int am = 0; am < 2; ++am)
#pragma unroll
            for (int bn = 0; bn < 4; ++bn)
                acc[am][bn] = __builtin_amdgcn_wmma_f32_16x16x32_bf16(
                    false, af[am].v, false, bf[bn].v,
                    (short)0, acc[am][bn], false, false);

#if USE_ASYNC
        __builtin_amdgcn_s_wait_asynccnt(0);
#endif
        __syncthreads();
        buf ^= 1;
    }

    // --- fused epilogue; C/D layout: VGPR e -> row e+8*sel, col = lane&15 ---
    int   ccol[4];
    float bcol[4];
#pragma unroll
    for (int bn = 0; bn < 4; ++bn) {
        ccol[bn] = cb + wn * 64 + bn * 16 + mrow;
        if constexpr (EPI != 2) bcol[bn] = bias[ccol[bn]];
    }
#pragma unroll
    for (int am = 0; am < 2; ++am)
#pragma unroll
        for (int e = 0; e < 8; ++e) {
            const int r = rb + wm * 32 + am * 16 + e + 8 * sel;
            float s = 0.f;
            if constexpr (EPI == 1 || EPI == 2 || EPI == 3) s = dinv[r];
            const size_t rowo = (size_t)r * ldd;
#pragma unroll
            for (int bn = 0; bn < 4; ++bn) {
                const float  v = acc[am][bn][e];
                const size_t o = rowo + ccol[bn];
                if constexpr (EPI == 1) {
                    // layer-1 GCN: relu(dinv*acc + 2*dinv^2*W1 + b1) -> bf16 h1
                    const float val =
                        fmaxf(fmaf(2.f * s * s, Mdiag[o], fmaf(s, v, bcol[bn])), 0.f);
                    outBF[o] = f2bf(val);
                } else if constexpr (EPI == 2) {
                    // M2 = h1@W2 (fp32, kept for diagonal term) + B2 = bf16(dinv*M2)
                    outF[o]  = v;
                    outBF[o] = f2bf(s * v);
                } else if constexpr (EPI == 3) {
                    // layer-2 GCN: h2 fp32 -> d_out, plus bf16 copy for the head
                    const float val =
                        fmaxf(fmaf(2.f * s * s, Mdiag[o], fmaf(s, v, bcol[bn])), 0.f);
                    outF[o]  = val;
                    outBF[o] = f2bf(val);
                } else {
                    // Z = relu(h2@Wf1 + bf1) (fp32)
                    outF[o] = fmaxf(v + bcol[bn], 0.f);
                }
            }
        }
}

// A[r] = dot(Z[r,:], Wf2) + bf2 ; one wave32 per row, shuffle reduction.
__global__ __launch_bounds__(256)
void k_rowdot(const float* __restrict__ Z, const float* __restrict__ Wf2,
              const float* __restrict__ bf2, float* __restrict__ outA,
              int rows, int cols) {
    const int wave = threadIdx.x >> 5, lane = threadIdx.x & 31;
    const int row = blockIdx.x * 8 + wave;
    if (row >= rows) return;
    const float* zr = Z + (size_t)row * cols;
    float s = 0.f;
    for (int c = lane; c < cols; c += 32) s += zr[c] * Wf2[c];
#pragma unroll
    for (int off = 16; off > 0; off >>= 1) s += __shfl_xor(s, off, 32);
    if (lane == 0) outA[row] = s + bf2[0];
}

// ---------------------------------------------------------------------------
extern "C" void kernel_launch(void* const* d_in, const int* in_sizes, int n_in,
                              void* d_out, int out_size, void* d_ws, size_t ws_size,
                              hipStream_t stream) {
    (void)in_sizes; (void)n_in; (void)out_size; (void)ws_size;

    const float* adj = (const float*)d_in[0];
    const float* W1  = (const float*)d_in[1];
    const float* b1  = (const float*)d_in[2];
    const float* W2  = (const float*)d_in[3];
    const float* b2  = (const float*)d_in[4];
    const float* Wf1 = (const float*)d_in[5];
    const float* bf1 = (const float*)d_in[6];
    const float* Wf2 = (const float*)d_in[7];
    const float* bf2 = (const float*)d_in[8];

    constexpr int Nn = 8192, H1 = 512, H2 = 128, H3 = 256;

    char* ws = (char*)d_ws;
    size_t off = 0;
    auto take = [&](size_t bytes) -> char* {
        off = (off + 255) & ~(size_t)255;
        char* p = ws + off;
        off += bytes;
        return p;
    };

    float* dinv  = (float*)take(sizeof(float) * Nn);
    u16*   adjb  = (u16*)  take(sizeof(u16) * (size_t)Nn * Nn);   // 128 MB
    u16*   B1    = (u16*)  take(sizeof(u16) * (size_t)Nn * H1);   // dinv*W1
    u16*   h1bf  = (u16*)  take(sizeof(u16) * (size_t)Nn * H1);
    float* M2    = (float*)take(sizeof(float) * (size_t)Nn * H2); // h1@W2
    u16*   B2    = (u16*)  take(sizeof(u16) * (size_t)Nn * H2);   // dinv*M2
    u16*   h2bf  = (u16*)  take(sizeof(u16) * (size_t)Nn * H2);
    float* Zf    = (float*)take(sizeof(float) * (size_t)Nn * H3);
    u16*   W2bf  = (u16*)  take(sizeof(u16) * H1 * H2);
    u16*   Wf1bf = (u16*)  take(sizeof(u16) * H2 * H3);

    float* outF = (float*)d_out;                 // h2 [8192*128] then A [8192]

    // 1) degrees + bf16 binarized adjacency (single pass over 256 MB)
    k_deg_binarize<<<Nn, 256, 0, stream>>>(adj, adjb, dinv, Nn);

    // 2) operand conversions
    k_scale_rows_bf16<<<4096, 256, 0, stream>>>(W1, dinv, B1, Nn, H1);
    k_f32_to_bf16<<<256, 256, 0, stream>>>(W2, W2bf, H1 * H2);
    k_f32_to_bf16<<<64, 256, 0, stream>>>(Wf1, Wf1bf, H2 * H3);

    // 3) GEMM1: h1 = relu(dinv*(adjb@B1) + 2dinv^2*W1 + b1)   (M=8192,N=512,K=8192)
    k_gemm_bf16<1><<<dim3(H1 / BN, Nn / BM), 256, 0, stream>>>(
        adjb, B1, nullptr, h1bf, dinv, W1, b1, Nn, Nn, H1, H1);

    // 4) GEMM2: M2 = h1@W2 (fp32) and B2 = bf16(dinv*M2)      (M=8192,N=128,K=512)
    k_gemm_bf16<2><<<dim3(H2 / BN, Nn / BM), 256, 0, stream>>>(
        h1bf, W2bf, M2, B2, dinv, nullptr, nullptr, H1, H1, H2, H2);

    // 5) GEMM3: h2 = relu(dinv*(adjb@B2) + 2dinv^2*M2 + b2) -> d_out + bf16 copy
    k_gemm_bf16<3><<<dim3(H2 / BN, Nn / BM), 256, 0, stream>>>(
        adjb, B2, outF, h2bf, dinv, M2, b2, Nn, Nn, H2, H2);

    // 6) GEMM4: Z = relu(h2@Wf1 + bf1)                        (M=8192,N=256,K=128)
    k_gemm_bf16<4><<<dim3(H3 / BN, Nn / BM), 256, 0, stream>>>(
        h2bf, Wf1bf, Zf, nullptr, nullptr, nullptr, bf1, H2, H2, H3, H3);

    // 7) A = Z@Wf2 + bf2 -> tail of d_out
    k_rowdot<<<Nn / 8, 256, 0, stream>>>(Zf, Wf2, bf2, outF + (size_t)Nn * H2, Nn, H3);
}
